// mLSTMCell_4054449127612
// MI455X (gfx1250) — matrix operand order repaired and checked by reading, not compiled, not run on hardware
//
#include <hip/hip_runtime.h>
#include <hip/hip_bf16.h>

#define S_LEN 2048
#define EMB_D 1024
#define NHEAD 8
#define HDIM  128
#define BM    64
#define BN    64
#define BATCH 2

// padded LDS strides (elements). 136*2B=272B and 72*2B=144B are both 16B-aligned
// and give conflict-free bank rotation (bank step 4 resp. 36 dwords per row).
#define QS_STRIDE 136
#define KS_STRIDE 136
#define VT_STRIDE 72
#define P_STRIDE  72

typedef __attribute__((ext_vector_type(16))) __bf16 v16bf;
typedef __attribute__((ext_vector_type(8)))  __bf16 v8bf;
typedef __attribute__((ext_vector_type(8)))  float  v8f;

__device__ __forceinline__ float log_sigmoid(float x) {
  return fminf(x, 0.f) - log1pf(expf(-fabsf(x)));
}

// -------- Kernel 1: gate preactivations  ig = x@Wi + bi ; lfg = logsigmoid(x@Wf + bf)
__global__ __launch_bounds__(256) void gates_kernel(
    const float* __restrict__ q, const float* __restrict__ k, const float* __restrict__ v,
    const float* __restrict__ igk, const float* __restrict__ igb,
    const float* __restrict__ fgk, const float* __restrict__ fgb,
    float* __restrict__ igbuf, float* __restrict__ lfgbuf) {
  const int row = blockIdx.x;               // b*S + s
  const int b = row / S_LEN, s = row % S_LEN;
  const int tid = threadIdx.x;
  float accI[NHEAD], accF[NHEAD];
#pragma unroll
  for (int hh = 0; hh < NHEAD; ++hh) { accI[hh] = 0.f; accF[hh] = 0.f; }

  for (int d = tid; d < 3 * EMB_D; d += 256) {
    const float* src = (d < EMB_D) ? q : ((d < 2 * EMB_D) ? k : v);
    const int dd = d & (EMB_D - 1);
    const float xv = src[(size_t)row * EMB_D + dd];
#pragma unroll
    for (int hh = 0; hh < NHEAD; ++hh) {
      accI[hh] = fmaf(xv, igk[d * NHEAD + hh], accI[hh]);
      accF[hh] = fmaf(xv, fgk[d * NHEAD + hh], accF[hh]);
    }
  }

  __shared__ float red[256];
  for (int hh = 0; hh < NHEAD; ++hh) {
    red[tid] = accI[hh]; __syncthreads();
    for (int off = 128; off > 0; off >>= 1) {
      if (tid < off) red[tid] += red[tid + off];
      __syncthreads();
    }
    if (tid == 0)
      igbuf[(size_t)(b * NHEAD + hh) * S_LEN + s] = red[0] + igb[hh];
    __syncthreads();

    red[tid] = accF[hh]; __syncthreads();
    for (int off = 128; off > 0; off >>= 1) {
      if (tid < off) red[tid] += red[tid + off];
      __syncthreads();
    }
    if (tid == 0)
      lfgbuf[(size_t)(b * NHEAD + hh) * S_LEN + s] = log_sigmoid(red[0] + fgb[hh]);
    __syncthreads();
  }
}

// -------- Kernel 2: inclusive cumsum of log-sigmoid forget gate per (b,h)
__global__ void cumsum_kernel(const float* __restrict__ lfg, float* __restrict__ csb) {
  const int bh = blockIdx.x;
  if (threadIdx.x == 0) {
    const float* src = lfg + (size_t)bh * S_LEN;
    float* dst = csb + (size_t)bh * S_LEN;
    float run = 0.f;
    for (int s = 0; s < S_LEN; ++s) { run += src[s]; dst[s] = run; }
  }
}

// ---- WMMA fragment loaders (contiguous k = 16*hv + s order, wave32), all-contiguous b128 ----
// A 16x32 bf16 from row-major [row][k]: lane row = nl,
// K in {hv*16 .. +15}
__device__ __forceinline__ v16bf load_a(const __bf16* base, int stride, int koff,
                                        int hv, int nl) {
  union { v16bf f; v8bf h[2]; } u;
  const __bf16* p = base + nl * stride + koff + hv * 16;
  u.h[0] = *(const v8bf*)p;
  u.h[1] = *(const v8bf*)(p + 8);
  return u.f;
}

// B 32x16 bf16 from [n][k] storage (k contiguous): lane col = n,
// lanes0-15 hold K=kbase..kbase+15, lanes16-31 K=kbase+16..kbase+31
__device__ __forceinline__ v16bf load_bk(const __bf16* base, int stride, int kbase,
                                         int n, int hv) {
  union { v16bf f; v8bf h[2]; } u;
  const __bf16* p = base + n * stride + kbase + hv * 16;
  u.h[0] = *(const v8bf*)p;
  u.h[1] = *(const v8bf*)(p + 8);
  return u.f;
}

// stage one K/V block (64 keys x 128 dh) into LDS: K row-major, V transposed
__device__ __forceinline__ void stage_kv(const float* __restrict__ kk,
                                         const float* __restrict__ vv,
                                         __bf16* KsB, __bf16* VTB,
                                         int b, int h, int jb, int tid) {
  for (int idx = tid * 4; idx < BN * HDIM; idx += 128 * 4) {
    const int r = idx >> 7, d = idx & (HDIM - 1);
    const size_t g = (size_t)(b * S_LEN + jb * BN + r) * EMB_D + h * HDIM + d;
    const float4 kv4 = *(const float4*)&kk[g];
    const float4 vv4 = *(const float4*)&vv[g];
    __bf16* kd = &KsB[r * KS_STRIDE + d];
    kd[0] = (__bf16)kv4.x; kd[1] = (__bf16)kv4.y;
    kd[2] = (__bf16)kv4.z; kd[3] = (__bf16)kv4.w;
    VTB[(d + 0) * VT_STRIDE + r] = (__bf16)vv4.x;
    VTB[(d + 1) * VT_STRIDE + r] = (__bf16)vv4.y;
    VTB[(d + 2) * VT_STRIDE + r] = (__bf16)vv4.z;
    VTB[(d + 3) * VT_STRIDE + r] = (__bf16)vv4.w;
  }
}

// -------- Kernel 3: streaming (flash-style) mLSTM head + RMSNorm
__global__ __launch_bounds__(128) void mlstm_kernel(
    const float* __restrict__ q, const float* __restrict__ kk, const float* __restrict__ vv,
    const float* __restrict__ igbuf, const float* __restrict__ csbuf,
    float* __restrict__ out) {
  __shared__ __bf16 Qs[BM * QS_STRIDE];          // scaled Q tile, row-major [row][dh]
  __shared__ __bf16 Ks[2][BN * KS_STRIDE];       // double-buffered K block [key][dh]
  __shared__ __bf16 VT[2][HDIM * VT_STRIDE];     // double-buffered V block [dh][key]
  __shared__ __bf16 Pbuf[4 * 16 * P_STRIDE];     // per-wave P strips [row][key]

  const int bh = blockIdx.y;
  const int b = bh / NHEAD, h = bh % NHEAD;
  const int tileq = blockIdx.x;
  const int tid = threadIdx.x;
  const int wave = tid >> 5, lane = tid & 31;
  const int hv = lane >> 4, nl = lane & 15;
  const int qrow0 = tileq * BM;
  const float kScale = 0.0883883476483184f;  // 1/sqrt(HD)

  // stage Q (scaled) into LDS as bf16 (float4 coalesced)
  for (int idx = tid * 4; idx < BM * HDIM; idx += 128 * 4) {
    const int r = idx >> 7, d = idx & (HDIM - 1);
    const float4 qv = *(const float4*)&q[(size_t)(b * S_LEN + qrow0 + r) * EMB_D + h * HDIM + d];
    __bf16* dst = &Qs[r * QS_STRIDE + d];
    dst[0] = (__bf16)(qv.x * kScale);
    dst[1] = (__bf16)(qv.y * kScale);
    dst[2] = (__bf16)(qv.z * kScale);
    dst[3] = (__bf16)(qv.w * kScale);
  }
  stage_kv(kk, vv, Ks[0], VT[0], b, h, 0, tid);
  __syncthreads();

  const float* cs = csbuf + (size_t)bh * S_LEN;
  const float* ig = igbuf + (size_t)bh * S_LEN;
  const int m0 = wave * 16;
  __bf16* Pw = &Pbuf[wave * 16 * P_STRIDE];

  float rcs[8], Mrow[8], Ssum[8];
  v8f Oacc[8];
#pragma unroll
  for (int i = 0; i < 8; ++i) {
    rcs[i] = cs[qrow0 + m0 + 8 * hv + i];
    Mrow[i] = -3.0e38f;
    Ssum[i] = 0.f;
  }
#pragma unroll
  for (int t = 0; t < 8; ++t) Oacc[t] = (v8f){0.f,0.f,0.f,0.f,0.f,0.f,0.f,0.f};

  v16bf aq[4];
#pragma unroll
  for (int c = 0; c < 4; ++c) aq[c] = load_a(&Qs[m0 * QS_STRIDE], QS_STRIDE, c * 32, hv, nl);

  for (int jb = 0; jb <= tileq; ++jb) {
    const __bf16* KsC = Ks[jb & 1];
    const __bf16* VTC = VT[jb & 1];

    // prefetch next block's K/V cachelines (128B granular) while we compute
    if (jb < tileq) {
      const size_t gnext = (size_t)(b * S_LEN + (jb + 1) * BN) * EMB_D + h * HDIM;
      for (int off = tid * 32; off < BN * HDIM; off += 128 * 32) {
        const int r = off >> 7, d = off & (HDIM - 1);
        __builtin_prefetch(&kk[gnext + (size_t)r * EMB_D + d], 0, 1);
        __builtin_prefetch(&vv[gnext + (size_t)r * EMB_D + d], 0, 1);
      }
    }

    // S tile = Q_strip @ K_block^T  (16 x 64, f32 accum), pipelined B loads
    v8f qkacc[4];
    {
      v16bf bcur = load_bk(KsC, KS_STRIDE, 0, nl, hv);
#pragma unroll
      for (int t = 0; t < 4; ++t) {
        v8f acc = (v8f){0.f,0.f,0.f,0.f,0.f,0.f,0.f,0.f};
#pragma unroll
        for (int c = 0; c < 4; ++c) {
          const int ni = t * 4 + c + 1;
          v16bf bnext = bcur;
          if (ni < 16)
            bnext = load_bk(KsC, KS_STRIDE, (ni & 3) * 32, (ni >> 2) * 16 + nl, hv);
          acc = __builtin_amdgcn_wmma_f32_16x16x32_bf16(false, aq[c], false, bcur,
                                                        (short)0, acc, false, false);
          bcur = bnext;
        }
        qkacc[t] = acc;
      }
    }

    // decay matrix values + running-max update
    float logD[4][8], tmax[8];
#pragma unroll
    for (int i = 0; i < 8; ++i) tmax[i] = -1.0e30f;
    const int mg0 = qrow0 + m0 + 8 * hv;
#pragma unroll
    for (int t = 0; t < 4; ++t) {
      const int jg = jb * BN + t * 16 + nl;
      const float ckd = ig[jg] - cs[jg];
#pragma unroll
      for (int i = 0; i < 8; ++i) {
        float val = rcs[i] + ckd;
        if (jg > mg0 + i) val = -1.0e30f;   // causal mask
        logD[t][i] = val;
        tmax[i] = fmaxf(tmax[i], val);
      }
    }
#pragma unroll
    for (int off = 1; off < 16; off <<= 1)
#pragma unroll
      for (int i = 0; i < 8; ++i)
        tmax[i] = fmaxf(tmax[i], __shfl_xor(tmax[i], off, 32));

    float scl[8];
#pragma unroll
    for (int i = 0; i < 8; ++i) {
      const float nm = fmaxf(Mrow[i], tmax[i]);
      scl[i] = __expf(Mrow[i] - nm);
      Mrow[i] = nm;
      Ssum[i] *= scl[i];
    }
#pragma unroll
    for (int t = 0; t < 8; ++t)
#pragma unroll
      for (int i = 0; i < 8; ++i) Oacc[t][i] *= scl[i];

    // P = qk * exp(logD - M); row sums; spill P to LDS for A-fragment reload
    float psum[8];
#pragma unroll
    for (int i = 0; i < 8; ++i) psum[i] = 0.f;
#pragma unroll
    for (int t = 0; t < 4; ++t)
#pragma unroll
      for (int i = 0; i < 8; ++i) {
        const float p = qkacc[t][i] * __expf(logD[t][i] - Mrow[i]);
        psum[i] += p;
        Pw[(i + 8 * hv) * P_STRIDE + t * 16 + nl] = (__bf16)p;
      }
#pragma unroll
    for (int off = 1; off < 16; off <<= 1)
#pragma unroll
      for (int i = 0; i < 8; ++i) psum[i] += __shfl_xor(psum[i], off, 32);
#pragma unroll
    for (int i = 0; i < 8; ++i) Ssum[i] += psum[i];

    // O += P @ V_block  (16 x 128), pipelined B loads
    {
      v16bf ap0 = load_a(Pw, P_STRIDE, 0, hv, nl);
      v16bf ap1 = load_a(Pw, P_STRIDE, 32, hv, nl);
      v16bf bcur = load_bk(VTC, VT_STRIDE, 0, nl, hv);
#pragma unroll
      for (int idx = 0; idx < 16; ++idx) {
        const int c = idx >> 3, t = idx & 7;
        v16bf bnext = bcur;
        if (idx < 15)
          bnext = load_bk(VTC, VT_STRIDE, ((idx + 1) >> 3) * 32, ((idx + 1) & 7) * 16 + nl, hv);
        Oacc[t] = __builtin_amdgcn_wmma_f32_16x16x32_bf16(false, c ? ap1 : ap0, false, bcur,
                                                          (short)0, Oacc[t], false, false);
        bcur = bnext;
      }
    }

    // stage next block into the other buffer (lines are hot from the prefetch)
    if (jb < tileq)
      stage_kv(kk, vv, Ks[(jb + 1) & 1], VT[(jb + 1) & 1], b, h, jb + 1, tid);
    __syncthreads();
  }

  // normalizer + RMSNorm over head dim, then store
  float rnorm[8], ms[8];
#pragma unroll
  for (int i = 0; i < 8; ++i) {
    const float nrm = fmaxf(fabsf(Ssum[i]), __expf(-Mrow[i]));
    rnorm[i] = 1.f / (nrm + 1e-6f);
    ms[i] = 0.f;
  }
#pragma unroll
  for (int t = 0; t < 8; ++t)
#pragma unroll
    for (int i = 0; i < 8; ++i) {
      const float o = Oacc[t][i] * rnorm[i];
      Oacc[t][i] = o;
      ms[i] = fmaf(o, o, ms[i]);
    }
#pragma unroll
  for (int off = 1; off < 16; off <<= 1)
#pragma unroll
    for (int i = 0; i < 8; ++i) ms[i] += __shfl_xor(ms[i], off, 32);
#pragma unroll
  for (int i = 0; i < 8; ++i) ms[i] = rsqrtf(ms[i] * (1.f / HDIM) + 1e-6f);

#pragma unroll
  for (int t = 0; t < 8; ++t)
#pragma unroll
    for (int i = 0; i < 8; ++i) {
      const int m = qrow0 + m0 + 8 * hv + i;
      out[(size_t)(b * S_LEN + m) * EMB_D + h * HDIM + t * 16 + nl] = Oacc[t][i] * ms[i];
    }
}

extern "C" void kernel_launch(void* const* d_in, const int* in_sizes, int n_in,
                              void* d_out, int out_size, void* d_ws, size_t ws_size,
                              hipStream_t stream) {
  const float* q   = (const float*)d_in[0];
  const float* k   = (const float*)d_in[1];
  const float* v   = (const float*)d_in[2];
  const float* igk = (const float*)d_in[3];
  const float* igb = (const float*)d_in[4];
  const float* fgk = (const float*)d_in[5];
  const float* fgb = (const float*)d_in[6];
  float* outp = (float*)d_out;

  const size_t n_bhs = (size_t)BATCH * NHEAD * S_LEN;  // 32768
  float* igbuf = (float*)d_ws;
  float* lfg   = igbuf + n_bhs;
  float* csb   = lfg + n_bhs;

  gates_kernel<<<dim3(BATCH * S_LEN), dim3(256), 0, stream>>>(
      q, k, v, igk, igb, fgk, fgb, igbuf, lfg);
  cumsum_kernel<<<dim3(BATCH * NHEAD), dim3(32), 0, stream>>>(lfg, csb);
  mlstm_kernel<<<dim3(S_LEN / BM, BATCH * NHEAD), dim3(128), 0, stream>>>(
      q, k, v, igbuf, csb, outp);
}
